// CrossTransFormer_86526411145611
// MI455X (gfx1250) — matrix operand order repaired and checked by reading, not compiled, not run on hardware
//
#include <hip/hip_runtime.h>
#include <hip/hip_bf16.h>

// ---------------------------------------------------------------------------
// Problem constants (from reference): B=8, C=1024, G=4, cg=256, PK=1024, PQ=4096
// ---------------------------------------------------------------------------
#define BATCH 8
#define CDIM  1024
#define GRP   4
#define CG    256
#define PK    1024
#define PQ    4096

#define BM 128      // workgroup M tile
#define BN 64       // workgroup N tile
#define KC 32       // k-step (one bf16 WMMA K)
#define KS 40       // LDS row stride in halves (pad 8; 80B keeps 16B alignment)
#define NTHREADS 256

typedef __attribute__((ext_vector_type(16))) __bf16 v16bf;
typedef __attribute__((ext_vector_type(8)))  __bf16 v8bf;
typedef __attribute__((ext_vector_type(4)))  __bf16 v4bf;
typedef __attribute__((ext_vector_type(8)))  float  v8f;
typedef __attribute__((ext_vector_type(4)))  int    v4i;

// CDNA5 async global->LDS copy (ASYNCcnt path); fallback = VGPR round-trip.
#if defined(__has_builtin)
#if __has_builtin(__builtin_amdgcn_global_load_async_to_lds_b128) && \
    __has_builtin(__builtin_amdgcn_s_wait_asynccnt)
#define USE_ASYNC_LDS 1
#endif
#endif
#ifndef USE_ASYNC_LDS
#define USE_ASYNC_LDS 0
#endif

__device__ inline void copy16_g2l(const void* gsrc, void* ldst) {
#if USE_ASYNC_LDS
    __builtin_amdgcn_global_load_async_to_lds_b128(
        (__attribute__((address_space(1))) v4i*)gsrc,
        (__attribute__((address_space(3))) v4i*)ldst, 0, 0);
#else
    *(uint4*)ldst = *(const uint4*)gsrc;
#endif
}

__device__ inline void copy16_fence() {
#if USE_ASYNC_LDS
    __builtin_amdgcn_s_wait_asynccnt(0);
#endif
}

// A-fragment: 16x32 bf16, CDNA5 wave32 layout.
// lane<16 : M=lane,    halves 0..7 -> K 0..7,   halves 8..15 -> K 16..23
// lane>=16: M=lane-16, halves 0..7 -> K 8..15,  halves 8..15 -> K 24..31
__device__ inline v16bf load_a_frag(const __bf16* rowBase /*&As[row][0]*/, int lHi) {
    v8bf lo = *(const v8bf*)(rowBase + lHi * 8);
    v8bf hi = *(const v8bf*)(rowBase + 16 + lHi * 8);
    v16bf a;
#pragma unroll
    for (int i = 0; i < 8; ++i) { a[i] = lo[i]; a[i + 8] = hi[i]; }
    return a;
}

// B-fragment: 32x16 bf16 staged transposed in LDS as BT[n][k].
// lane<16 : N=lane,    halves 0..7 -> K 0..7,   halves 8..15 -> K 16..23
// lane>=16: N=lane-16, halves 0..7 -> K 8..15,  halves 8..15 -> K 24..31
__device__ inline v16bf load_b_frag(const __bf16* rowBase /*&Bs[col][0]*/, int lHi) {
    v8bf lo = *(const v8bf*)(rowBase + lHi * 8);
    v8bf hi = *(const v8bf*)(rowBase + 16 + lHi * 8);
    v16bf b;
#pragma unroll
    for (int i = 0; i < 8; ++i) { b[i] = lo[i]; b[i + 8] = hi[i]; }
    return b;
}

// AM: 0 = A is f32 row-major [M,K]; 1 = A is bf16 row-major [M,K]
// BMD: 0 = B is f32 [N,K] (direct); 1 = B is bf16 [N,K] (direct);
//      2 = B is f32 [K,N] (transpose during staging)
// EP:  0 = bf16 store natural  Out[c,n] = acc + bias[c]
//      1 = bf16 store transposed Out[n,c] = acc + bias[c]
//      2 = f32 store Out[c,n] = acc * scale
//      3 = loss: sum((acc - I)^2) -> atomicAdd(lossPtr, * lossScale)
template <int AM, int BMD, int EP>
__global__ __launch_bounds__(NTHREADS)
void gemm_tile_kernel(const void* __restrict__ Aglob, long aBatchStride, long aGroupStride, int lda,
                      const void* __restrict__ Bglob, long bBatchStride, long bGroupStride, int ldb,
                      void* __restrict__ Out, long oBatchStride, int ldo,
                      const float* __restrict__ bias,
                      float scale, float* __restrict__ lossPtr, float lossScale,
                      int Kdim, int mTilesPerGroup, int groupChanStride) {
    __shared__ __align__(16) __bf16 As[BM][KS];
    __shared__ __align__(16) __bf16 Bs[BN][KS];
    __shared__ float red[NTHREADS];

    const int tid  = threadIdx.x;
    const int z    = blockIdx.z;
    const int g    = blockIdx.y / mTilesPerGroup;
    const int mt   = blockIdx.y % mTilesPerGroup;
    const int n0   = blockIdx.x * BN;
    const int row0 = mt * BM;

    const long aBase = z * aBatchStride + g * aGroupStride;
    const long bBase = z * bBatchStride + g * bGroupStride;

    const int lane = tid & 31;
    const int wave = tid >> 5;
    const int wm   = wave & 3;   // 4 waves along M -> 128
    const int wn   = wave >> 2;  // 2 waves along N -> 64
    const int l15  = lane & 15;
    const int lHi  = lane >> 4;

    v8f acc[2][2];
#pragma unroll
    for (int i = 0; i < 2; ++i)
#pragma unroll
        for (int j = 0; j < 2; ++j) acc[i][j] = (v8f)(0.0f);

    for (int kb = 0; kb < Kdim; kb += KC) {
        __syncthreads();
        // ---- stage A tile (BM x KC) ----
        if (AM == 0) {
            // f32 -> bf16, vectorized: float4 loads, ds_store_b64 stores
            const float* A = (const float*)Aglob + aBase;
#pragma unroll
            for (int it = 0; it < (BM * (KC / 4)) / NTHREADS; ++it) {
                int idx = tid + it * NTHREADS;
                int r = idx >> 3, kq = idx & 7;
                const float4 f = *(const float4*)(A + (long)(row0 + r) * lda + kb + kq * 4);
                v4bf o;
                o[0] = (__bf16)f.x; o[1] = (__bf16)f.y;
                o[2] = (__bf16)f.z; o[3] = (__bf16)f.w;
                *(v4bf*)&As[r][kq * 4] = o;
            }
        } else {
            // bf16 direct: async global->LDS 16B transfers
            const __bf16* A = (const __bf16*)Aglob + aBase;
#pragma unroll
            for (int it = 0; it < (BM * 4) / NTHREADS; ++it) {
                int idx = tid + it * NTHREADS;
                int r = idx >> 2, kd = idx & 3;
                copy16_g2l(A + (long)(row0 + r) * lda + kb + kd * 8,
                           (uint4*)&As[r][0] + kd);
            }
        }
        // ---- stage B tile as BT[n][k] (BN x KC) ----
        if (BMD == 0) {
            const float* Bsrc = (const float*)Bglob + bBase;
#pragma unroll
            for (int it = 0; it < (BN * (KC / 4)) / NTHREADS; ++it) {
                int idx = tid + it * NTHREADS;
                int r = idx >> 3, kq = idx & 7;
                const float4 f = *(const float4*)(Bsrc + (long)(n0 + r) * ldb + kb + kq * 4);
                v4bf o;
                o[0] = (__bf16)f.x; o[1] = (__bf16)f.y;
                o[2] = (__bf16)f.z; o[3] = (__bf16)f.w;
                *(v4bf*)&Bs[r][kq * 4] = o;
            }
        } else if (BMD == 1) {
            const __bf16* Bsrc = (const __bf16*)Bglob + bBase;
            {
                int idx = tid;  // BN*4 == 256 == NTHREADS
                int r = idx >> 2, kd = idx & 3;
                copy16_g2l(Bsrc + (long)(n0 + r) * ldb + kb + kd * 8,
                           (uint4*)&Bs[r][0] + kd);
            }
        } else {
            // f32 [K,N] transpose: float4 loads along n (coalesced), scattered b16 stores
            const float* Bsrc = (const float*)Bglob + bBase;
#pragma unroll
            for (int it = 0; it < (KC * (BN / 4)) / NTHREADS; ++it) {
                int idx = tid + it * NTHREADS;
                int k = idx >> 4, nq = idx & 15;
                const float4 f = *(const float4*)(Bsrc + (long)(kb + k) * ldb + n0 + nq * 4);
                Bs[nq * 4 + 0][k] = (__bf16)f.x;
                Bs[nq * 4 + 1][k] = (__bf16)f.y;
                Bs[nq * 4 + 2][k] = (__bf16)f.z;
                Bs[nq * 4 + 3][k] = (__bf16)f.w;
            }
        }
        if (AM == 1 || BMD == 1) copy16_fence();
        __syncthreads();

        // ---- compute: 2x2 WMMA frags per wave ----
        v16bf afr[2], bfr[2];
#pragma unroll
        for (int fm = 0; fm < 2; ++fm)
            afr[fm] = load_a_frag(&As[wm * 32 + fm * 16 + l15][0], lHi);
#pragma unroll
        for (int fn = 0; fn < 2; ++fn)
            bfr[fn] = load_b_frag(&Bs[wn * 32 + fn * 16 + l15][0], lHi);
#pragma unroll
        for (int fm = 0; fm < 2; ++fm)
#pragma unroll
            for (int fn = 0; fn < 2; ++fn)
                acc[fm][fn] = __builtin_amdgcn_wmma_f32_16x16x32_bf16(
                    false, afr[fm], false, bfr[fn], (short)0, acc[fm][fn],
                    false, false);
    }

    // ---- epilogue ----
    float lsum = 0.0f;
#pragma unroll
    for (int fm = 0; fm < 2; ++fm) {
#pragma unroll
        for (int fn = 0; fn < 2; ++fn) {
#pragma unroll
            for (int r = 0; r < 8; ++r) {
                // C/D layout: lane<16 -> M=r, N=lane ; lane>=16 -> M=8+r, N=lane-16
                int mm = wm * 32 + fm * 16 + r + lHi * 8;
                int nn = wn * 32 + fn * 16 + l15;
                float v = acc[fm][fn][r];
                int cIdx = g * groupChanStride + row0 + mm;
                int nG = n0 + nn;
                if (EP == 0) {
                    __bf16* O = (__bf16*)Out + z * oBatchStride;
                    O[(long)cIdx * ldo + nG] = (__bf16)(v + bias[cIdx]);
                } else if (EP == 1) {
                    __bf16* O = (__bf16*)Out + z * oBatchStride;
                    O[(long)nG * ldo + cIdx] = (__bf16)(v + bias[cIdx]);
                } else if (EP == 2) {
                    float* O = (float*)Out + z * oBatchStride;
                    O[(long)cIdx * ldo + nG] = v * scale;
                } else {
                    float d = v - ((cIdx == nG) ? 1.0f : 0.0f);
                    lsum += d * d;
                }
            }
        }
    }
    if (EP == 3) {
        red[tid] = lsum;
        __syncthreads();
#pragma unroll
        for (int s = NTHREADS / 2; s > 0; s >>= 1) {
            if (tid < s) red[tid] += red[tid + s];
            __syncthreads();
        }
        if (tid == 0) atomicAdd(lossPtr, red[0] * lossScale);
    }
}

// Column softmax over i (PK rows) for each (b, j): H[b, i, j], axis=-2.
__global__ __launch_bounds__(NTHREADS)
void softmax_col_kernel(float* __restrict__ H) {
    const int j = blockIdx.x * NTHREADS + threadIdx.x;
    const long base = (long)blockIdx.y * PK * PQ + j;
    float m = -INFINITY, s = 0.0f;
    for (int i = 0; i < PK; ++i) {
        float x = H[base + (long)i * PQ];
        float nm = fmaxf(m, x);
        s = s * __expf(m - nm) + __expf(x - nm);
        m = nm;
    }
    const float inv = 1.0f / s;
    for (int i = 0; i < PK; ++i) {
        long a = base + (long)i * PQ;
        H[a] = __expf(H[a] - m) * inv;
    }
}

__global__ void init_loss_kernel(float* loss) { *loss = 0.0f; }

extern "C" void kernel_launch(void* const* d_in, const int* in_sizes, int n_in,
                              void* d_out, int out_size, void* d_ws, size_t ws_size,
                              hipStream_t stream) {
    const float* Xq = (const float*)d_in[0];  // [B, C, PQ]
    const float* Xk = (const float*)d_in[1];  // [B, C, PK]
    const float* Wq = (const float*)d_in[2];  // [G, cg, cg]
    const float* bq = (const float*)d_in[3];  // [C]
    const float* Wk = (const float*)d_in[4];
    const float* bk = (const float*)d_in[5];
    const float* Wv = (const float*)d_in[6];
    const float* bv = (const float*)d_in[7];

    // output layout: out [B,C,PQ] f32 | loss [1] | H [B,PK,PQ] f32
    float* outP  = (float*)d_out;
    float* lossP = outP + (size_t)BATCH * CDIM * PQ;
    float* Hp    = lossP + 1;

    // workspace: Qbf [B,C,PK] | Ktb [B,PK,C] | Vtb [B,PQ,C]  (all bf16)
    __bf16* Qbf = (__bf16*)d_ws;
    __bf16* Ktb = Qbf + (size_t)BATCH * CDIM * PK;
    __bf16* Vtb = Ktb + (size_t)BATCH * PK * CDIM;

    const float SCALE = 0.03125f;  // 1/sqrt(1024)
    const float lossScale = 1.0f / ((float)BATCH * (float)PK * (float)PK);

    init_loss_kernel<<<1, 1, 0, stream>>>(lossP);

    // --- grouped projections: Y = W_g @ X_g + b  (A=W f32, B=X f32 transpose-staged)
    // Q = Wq @ Xk  -> natural bf16 [B, C, PK]
    gemm_tile_kernel<0, 2, 0><<<dim3(PK / BN, GRP * (CG / BM), BATCH), NTHREADS, 0, stream>>>(
        Wq, 0L, (long)CG * CG, CG,
        Xk, (long)CDIM * PK, (long)CG * PK, PK,
        Qbf, (long)CDIM * PK, PK,
        bq, 1.0f, nullptr, 0.0f, CG, CG / BM, CG);
    // K = Wk @ Xk  -> transposed bf16 Kt [B, PK, C]
    gemm_tile_kernel<0, 2, 1><<<dim3(PK / BN, GRP * (CG / BM), BATCH), NTHREADS, 0, stream>>>(
        Wk, 0L, (long)CG * CG, CG,
        Xk, (long)CDIM * PK, (long)CG * PK, PK,
        Ktb, (long)PK * CDIM, CDIM,
        bk, 1.0f, nullptr, 0.0f, CG, CG / BM, CG);
    // V = Wv @ Xq  -> transposed bf16 Vt [B, PQ, C]
    gemm_tile_kernel<0, 2, 1><<<dim3(PQ / BN, GRP * (CG / BM), BATCH), NTHREADS, 0, stream>>>(
        Wv, 0L, (long)CG * CG, CG,
        Xq, (long)CDIM * PQ, (long)CG * PQ, PQ,
        Vtb, (long)PQ * CDIM, CDIM,
        bv, 1.0f, nullptr, 0.0f, CG, CG / BM, CG);

    // --- H[b,i,j] = SCALE * sum_c Kt[b,i,c] * Vt[b,j,c]   (A,B direct bf16, async-LDS)
    gemm_tile_kernel<1, 1, 2><<<dim3(PQ / BN, PK / BM, BATCH), NTHREADS, 0, stream>>>(
        Ktb, (long)PK * CDIM, 0L, CDIM,
        Vtb, (long)PQ * CDIM, 0L, CDIM,
        Hp, (long)PK * PQ, PQ,
        nullptr, SCALE, nullptr, 0.0f, CDIM, PK / BM, 0);

    // --- softmax over PK (column-wise), in place on H
    softmax_col_kernel<<<dim3(PQ / NTHREADS, BATCH), NTHREADS, 0, stream>>>(Hp);

    // --- sparse loss: mid = H @ H^T over j; sum((mid - I)^2)/count
    gemm_tile_kernel<0, 0, 3><<<dim3(PK / BN, PK / BM, BATCH), NTHREADS, 0, stream>>>(
        Hp, (long)PK * PQ, 0L, PQ,
        Hp, (long)PK * PQ, 0L, PQ,
        nullptr, 0L, 0,
        nullptr, 1.0f, lossP, lossScale, PQ, PK / BM, 0);

    // --- out[b,c,j] = sum_i Q[b,c,i] * H[b,i,j]   (A bf16 direct, B f32 transpose)
    gemm_tile_kernel<1, 2, 2><<<dim3(PQ / BN, CDIM / BM, BATCH), NTHREADS, 0, stream>>>(
        Qbf, (long)CDIM * PK, 0L, PK,
        Hp, (long)PK * PQ, 0L, PQ,
        outP, (long)CDIM * PQ, PQ,
        nullptr, 1.0f, nullptr, 0.0f, PK, CDIM / BM, 0);
}
